// SpatialTemporalGNN_42210938585391
// MI455X (gfx1250) — matrix operand order repaired and checked by reading, not compiled, not run on hardware
//
#include <hip/hip_runtime.h>

#define NN 100000
#define MM 800000
#define KK 6
#define HH 64
#define NHD 4
#define DHH 16
#define NROWTILES 37500   // (NN*KK)/16

typedef _Float16 half_t;
typedef __attribute__((ext_vector_type(16))) _Float16 v16h;
typedef __attribute__((ext_vector_type(8)))  _Float16 v8h;
typedef __attribute__((ext_vector_type(4)))  _Float16 v4h;
typedef __attribute__((ext_vector_type(8)))  float    v8f;
typedef __attribute__((ext_vector_type(4)))  float    v4f;
typedef __attribute__((ext_vector_type(4)))  unsigned int v4u;
typedef __attribute__((ext_vector_type(4)))  int      v4i;
typedef __attribute__((ext_vector_type(8)))  int      v8i;

// ---------------------------------------------------------------- WMMA helpers
__device__ inline v8f wmma16(v16h a, v16h b, v8f c) {
  // v_wmma_f32_16x16x32_f16
  return __builtin_amdgcn_wmma_f32_16x16x32_f16(false, a, false, b, (short)0, c,
                                                false, false);
}

// A-fragment (16x32, f16) from LDS row-major [rows x lda].
// lane {half,m}: element e <-> row m, K = k0 + e + (e>=8?8:0) + half*8
// -> two contiguous 8-half runs => 2x ds_load_b128
__device__ inline v16h a_frag_lds(const half_t* src, int lda, int r0, int k0) {
  int lane = threadIdx.x & 31;
  int hf = lane >> 4, m = lane & 15;
  const half_t* p = &src[(r0 + m) * lda + k0 + hf * 8];
  v8h lo = *(const v8h*)p;
  v8h hi = *(const v8h*)(p + 16);
  v16h a;
#pragma unroll
  for (int e = 0; e < 8; ++e) { a[e] = lo[e]; a[e + 8] = hi[e]; }
  return a;
}

// B-fragment (32x16, f16) from fragment-ready transposed f16 weights:
// wT[n * KD + k]; lane {half,m}: element e <-> K = k0 + e + (e>=8?8:0) + half*8, col = n0+m
// -> two contiguous 8-half runs => 2x b128
__device__ inline v16h b_frag_t(const half_t* wT, int KD, int k0, int n0) {
  int lane = threadIdx.x & 31;
  int hf = lane >> 4, m = lane & 15;
  const half_t* p = wT + (n0 + m) * KD + k0 + hf * 8;
  v8h lo = *(const v8h*)p;
  v8h hi = *(const v8h*)(p + 16);
  v16h b;
#pragma unroll
  for (int e = 0; e < 8; ++e) { b[e] = lo[e]; b[e + 8] = hi[e]; }
  return b;
}

// 48 x KD (LDS f16) @ KD x ND (transposed f16 weights) + bias -> 48 x ND (LDS f16)
// B fragments hoisted across the 3 row tiles.
template <int KD, int ND>
__device__ inline void gemm48_t(const half_t* A, int lda, const half_t* wT,
                                const float* bias, half_t* out, int ldo, bool relu_) {
  int lane = threadIdx.x & 31;
  int hf = lane >> 4, m = lane & 15;
#pragma unroll
  for (int ct = 0; ct < ND / 16; ++ct) {
    v16h bf[KD / 32];
#pragma unroll
    for (int kc = 0; kc < KD / 32; ++kc) bf[kc] = b_frag_t(wT, KD, kc * 32, ct * 16);
    int col = ct * 16 + m;
    float bv = bias[col];
#pragma unroll
    for (int t = 0; t < 3; ++t) {
      int r0 = t * 16;
      v8f acc = {};
#pragma unroll
      for (int kc = 0; kc < KD / 32; ++kc)
        acc = wmma16(a_frag_lds(A, lda, r0, kc * 32), bf[kc], acc);
#pragma unroll
      for (int e = 0; e < 8; ++e) {
        float v = acc[e] + bv;
        if (relu_) v = v > 0.f ? v : 0.f;
        out[(r0 + e + hf * 8) * ldo + col] = (half_t)v;
      }
    }
  }
}

// ---------------------------------------------------------------- prep kernels
__global__ void k_zero_f32(float* p, int n) {
  int i = blockIdx.x * 256 + threadIdx.x;
  if (i < n) p[i] = 0.f;
}

__global__ void k_deg(const int* __restrict__ ei, float* deg) {
  int e = blockIdx.x * 256 + threadIdx.x;  // grid covers MM exactly
  atomicAdd(&deg[ei[MM + e]], 1.0f);
}

__global__ void k_dinv(float* deg) {
  int i = blockIdx.x * 256 + threadIdx.x;
  if (i < NN) deg[i] = rsqrtf(deg[i] + 1.0f);
}

// out[n*Kpad + k] = (k < Krows) ? (f16)W[k*Ncols + n] : 0   (fragment-ready f16)
__global__ void k_transp(const float* __restrict__ W, half_t* __restrict__ out,
                         int Krows, int Kpad, int Ncols) {
  int i = blockIdx.x * 256 + threadIdx.x;
  if (i >= Ncols * Kpad) return;
  int n = i / Kpad, k = i % Kpad;
  out[i] = (k < Krows) ? (half_t)W[k * Ncols + n] : (half_t)0.0f;
}

// ------------------------------------------------ proj (tanh) + GCN transform
__global__ void k_proj_gcn(const float* __restrict__ x, const float* __restrict__ u,
                           const float* __restrict__ u_seq,
                           const half_t* __restrict__ wpT,   // 64 x 32
                           const float* __restrict__ b_proj,
                           const half_t* __restrict__ wgT,   // 64 x 64
                           half_t* __restrict__ xg) {
  __shared__ half_t hsh[8][1024];  // per-wave 16x64 tile
  int wv = threadIdx.x >> 5;
  int lane = threadIdx.x & 31;
  int hf = lane >> 4, m = lane & 15;
  int T = blockIdx.x * 8 + wv;
  bool active = T < NROWTILES;
  if (!active) T = NROWTILES - 1;
  int r0 = T * 16;

  // build A (16x32) of node_in = [x(12)|u(11)|u_seq(5)|0] per the CDNA5 A layout
  v16h a;
#pragma unroll
  for (int e = 0; e < 16; ++e) {
    int k = e + (e >= 8 ? 8 : 0) + hf * 8;
    int row = r0 + m;
    int n = row / KK, s = row % KK;
    float v = 0.f;
    if (k < 12) v = x[n * 12 + k];
    else if (k < 23) v = u[k - 12];
    else if (k < 28) v = u_seq[s * 5 + (k - 23)];
    a[e] = (half_t)v;
  }
  half_t* hl = &hsh[wv][0];
#pragma unroll
  for (int ct = 0; ct < 4; ++ct) {
    v8f acc = {};
    acc = wmma16(a, b_frag_t(wpT, 32, 0, ct * 16), acc);
    int col = ct * 16 + m;
    float bp = b_proj[col];
#pragma unroll
    for (int e = 0; e < 8; ++e)
      hl[(e + hf * 8) * 64 + col] = (half_t)tanhf(acc[e] + bp);
  }
  __syncthreads();

  v16h a0 = a_frag_lds(hl, 64, 0, 0);
  v16h a1 = a_frag_lds(hl, 64, 0, 32);
#pragma unroll
  for (int ct = 0; ct < 4; ++ct) {
    v8f acc = {};
    acc = wmma16(a0, b_frag_t(wgT, 64, 0, ct * 16), acc);
    acc = wmma16(a1, b_frag_t(wgT, 64, 32, ct * 16), acc);
    int col = ct * 16 + m;
    if (active) {
#pragma unroll
      for (int e = 0; e < 8; ++e)
        xg[(size_t)(r0 + e + hf * 8) * 64 + col] = (half_t)acc[e];
    }
  }
}

// agg = xg * dinv[n]^2   (self-loop term), vectorized b64-load/b128-store
__global__ void k_init_agg(const half_t* __restrict__ xg,
                           const float* __restrict__ dinv, float* __restrict__ agg) {
  size_t i4 = (size_t)blockIdx.x * 256 + threadIdx.x;  // over N*K*64/4 exactly
  int n = (int)(i4 / 96);
  float d = dinv[n];
  d = d * d;
  v4h xv = *(const v4h*)(xg + i4 * 4);
  v4f o;
#pragma unroll
  for (int j = 0; j < 4; ++j) o[j] = (float)xv[j] * d;
  *(v4f*)(agg + i4 * 4) = o;
}

// one wave per edge: agg[dst] += xg[src] * dinv[src]*dinv[dst]  (K*H floats)
__global__ void k_scatter(const int* __restrict__ ei, const float* __restrict__ dinv,
                          const half_t* __restrict__ xg, float* __restrict__ agg) {
  int wv = threadIdx.x >> 5, lane = threadIdx.x & 31;
  int e = blockIdx.x * 8 + wv;  // grid covers MM exactly
  int sn = ei[e], dn = ei[MM + e];
  float nrm = dinv[sn] * dinv[dn];
  size_t sb = (size_t)sn * (KK * 64), db = (size_t)dn * (KK * 64);
  __builtin_prefetch(xg + sb, 0, 1);  // global_prefetch_b8
  int base = lane * 12;               // 12 contiguous f16 per lane
#pragma unroll
  for (int t = 0; t < 3; ++t) {
    v4h vv = *(const v4h*)(xg + sb + base + t * 4);
#pragma unroll
    for (int j = 0; j < 4; ++j)
      atomicAdd(&agg[db + base + t * 4 + j], (float)vv[j] * nrm);
  }
}

// ---------------------------------------------------- transformer (8 nodes/wave)
__device__ inline void layernorm48(half_t* H, const half_t* R, const float* g,
                                   const float* b) {
  int lane = threadIdx.x & 31;
  for (int row = lane; row < 48; row += 32) {
    float mean = 0.f, sq = 0.f;
    for (int c = 0; c < 64; ++c) {
      float z = (float)H[row * 64 + c] + (float)R[row * 64 + c];
      mean += z; sq += z * z;
    }
    mean *= (1.f / 64.f);
    float var = sq * (1.f / 64.f) - mean * mean;
    float inv = rsqrtf(var + 1e-5f);
    for (int c = 0; c < 64; ++c) {
      float z = (float)H[row * 64 + c] + (float)R[row * 64 + c];
      H[row * 64 + c] = (half_t)((z - mean) * inv * g[c] + b[c]);
    }
  }
}

__global__ void k_transformer(const float* __restrict__ agg, const float* __restrict__ b_gcn,
                              const half_t* wqT, const float* bq, const half_t* wkT,
                              const float* bk, const half_t* wvT, const float* bv,
                              const half_t* woT, const float* bo, const float* ln1g,
                              const float* ln1b, const half_t* wf1T, const float* bf1,
                              const half_t* wf2T, const float* bf2, const float* ln2g,
                              const float* ln2b, float* __restrict__ hfin) {
  __shared__ half_t sh[2][15872];  // 31 KB arena per wave
  int wv = threadIdx.x >> 5;
  int lane = threadIdx.x & 31;
  half_t* H  = &sh[wv][0];      // 48x64
  half_t* Q  = H + 3072;        // 48x64
  half_t* Kb = H + 6144;        // 48x64
  half_t* V  = H + 9216;        // 48x64
  half_t* AO = H + 12288;       // 48x64
  float*  SC = (float*)(H + 15360);  // 4x6x6 scores
  int nb = (blockIdx.x * 2 + wv) * 8;  // 8 nodes per wave, exact partition of N

  // H = relu(agg + b_gcn), vectorized float4 loads
  const v4f* bg4 = (const v4f*)b_gcn;
  for (int idx = lane; idx < 768; idx += 32) {
    int row = idx >> 4, c4 = idx & 15;
    int n = nb + row / KK, s = row % KK;
    v4f v = *(const v4f*)&agg[((size_t)n * KK + s) * 64 + c4 * 4];
    v4f bb = bg4[c4];
    v4h hv;
#pragma unroll
    for (int j = 0; j < 4; ++j) {
      float z = v[j] + bb[j];
      hv[j] = (half_t)(z > 0.f ? z : 0.f);
    }
    *(v4h*)&H[row * 64 + c4 * 4] = hv;
  }
  __syncthreads();

  gemm48_t<64, 64>(H, 64, wqT, bq, Q, 64, false);
  gemm48_t<64, 64>(H, 64, wkT, bk, Kb, 64, false);
  gemm48_t<64, 64>(H, 64, wvT, bv, V, 64, false);
  __syncthreads();

  // attention over K=6 steps, NH=4 heads of DH=16 (VALU; tiny)
  for (int ni = 0; ni < 8; ++ni) {
    int rb = ni * KK;
    for (int idx = lane; idx < NHD * KK * KK; idx += 32) {
      int hd = idx / 36, rem = idx % 36, i = rem / 6, j = rem % 6;
      float s = 0.f;
#pragma unroll
      for (int d = 0; d < DHH; ++d)
        s += (float)Q[(rb + i) * 64 + hd * DHH + d] *
             (float)Kb[(rb + j) * 64 + hd * DHH + d];
      SC[idx] = s * 0.25f;  // 1/sqrt(16)
    }
    __syncthreads();
    if (lane < NHD * KK) {
      int hd = lane / KK, i = lane % KK;
      float mx = -1e30f;
      for (int j = 0; j < KK; ++j) mx = fmaxf(mx, SC[hd * 36 + i * 6 + j]);
      float ex[KK], sum = 0.f;
      for (int j = 0; j < KK; ++j) {
        ex[j] = __expf(SC[hd * 36 + i * 6 + j] - mx);
        sum += ex[j];
      }
      float inv = 1.f / sum;
      for (int j = 0; j < KK; ++j) SC[hd * 36 + i * 6 + j] = ex[j] * inv;
    }
    __syncthreads();
    for (int idx = lane; idx < KK * 64; idx += 32) {
      int i = idx >> 6, c = idx & 63;
      int hd = c / DHH;
      float s = 0.f;
      for (int j = 0; j < KK; ++j)
        s += SC[hd * 36 + i * 6 + j] * (float)V[(rb + j) * 64 + c];
      AO[(rb + i) * 64 + c] = (half_t)s;
    }
    __syncthreads();
  }

  gemm48_t<64, 64>(AO, 64, woT, bo, Q, 64, false);  // O -> Q region
  __syncthreads();
  layernorm48(H, Q, ln1g, ln1b);                    // H = LN(H + O)
  __syncthreads();
  half_t* T = Kb;                                   // 48x128 over K/V regions
  gemm48_t<64, 128>(H, 64, wf1T, bf1, T, 128, true);
  __syncthreads();
  gemm48_t<128, 64>(T, 128, wf2T, bf2, Q, 64, false);
  __syncthreads();
  layernorm48(H, Q, ln2g, ln2b);                    // H = LN(H + FF)
  __syncthreads();

  // emit last time-step embedding per node (vectorized b64->b128)
  for (int idx = lane; idx < 128; idx += 32) {
    int ni = idx >> 4, c4 = idx & 15;
    v4h hv = *(const v4h*)&H[(ni * KK + KK - 1) * 64 + c4 * 4];
    v4f o;
#pragma unroll
    for (int j = 0; j < 4; ++j) o[j] = (float)hv[j];
    *(v4f*)&hfin[(size_t)(nb + ni) * 64 + c4 * 4] = o;
  }
}

// ---------------------------------------------------------------- edge MLP
__global__ void k_edge_mlp(const int* __restrict__ ei, const float* __restrict__ eattr,
                           const float* __restrict__ eseq, const float* __restrict__ hfin,
                           const half_t* __restrict__ wd1T,  // 64 x 128 (rows 0..127)
                           const float* __restrict__ Wd1,    // full 137 x 64 (tail rows)
                           const float* bd1,
                           const half_t* __restrict__ wd2T,  // 32 x 64
                           const float* bd2, const float* Wd3, const float* bd3,
                           float* __restrict__ out) {
  __shared__ half_t esh[4][3968];
  __shared__ half_t WL[8192 + 2048];  // Wd1T (TDM) + Wd2T
  int wv = threadIdx.x >> 5;
  int lane = threadIdx.x & 31;
  int hf = lane >> 4, m = lane & 15;
  half_t* HS = &esh[wv][0];          // 16x64 f16
  half_t* HD = HS + 1024;            // 16x64 f16
  half_t* Z1 = HS + 2048;            // 16x64 f16
  half_t* Z2 = HS + 3072;            // 16x32 f16
  float*  EX = (float*)(HS + 3584);  // 16x9 f32
  int*   IDs = (int*)(HS + 3872);    // 16 src + 16 dst
  int e0 = (blockIdx.x * 4 + wv) * 16;  // exact partition of M

  // --- stage W_d1^T into LDS via the Tensor Data Mover (wave 0), W_d2^T via ALU
#if __has_builtin(__builtin_amdgcn_tensor_load_to_lds)
  if (threadIdx.x < 32) {
    unsigned lds = (unsigned)(uintptr_t)(&WL[0]);
    unsigned long long ga = (unsigned long long)(uintptr_t)wd1T;
    // D# group0: count=1 | lds_addr | global_addr | type=2
    v4u g0 = {1u, lds, (unsigned)ga, (unsigned)(ga >> 32) | (2u << 30)};
    // D# group1: data_size=2B; 1-D tensor/tile of 8192 elements
    v8i g1 = {(int)(1u << 16),          // workgroup_mask=0, data_size=1 (2 bytes)
              (int)(8192u << 16),       // tensor_dim0[15:0] << 48
              (int)(1u << 16),          // tensor_dim0[31:16]=0, tensor_dim1=1
              (int)(8192u << 16),       // tile_dim0 = 8192
              1,                        // tile_dim1 = 1
              8192,                     // tensor_dim0_stride
              0, 0};
    v4i z4 = {};
#if __clang_major__ >= 23
    v8i z8 = {};
    __builtin_amdgcn_tensor_load_to_lds(g0, g1, z4, z4, z8, 0);
#else
    __builtin_amdgcn_tensor_load_to_lds(g0, g1, z4, z4, 0);
#endif
    __builtin_amdgcn_s_wait_tensorcnt(0);
  }
#else
  for (int idx = threadIdx.x; idx < 8192 / 8; idx += 128)
    ((v8h*)WL)[idx] = ((const v8h*)wd1T)[idx];
#endif
  half_t* WL2 = WL + 8192;
  for (int idx = threadIdx.x; idx < 2048 / 8; idx += 128)
    ((v8h*)WL2)[idx] = ((const v8h*)wd2T)[idx];

  if (lane < 16) {
    IDs[lane] = ei[e0 + lane];
    IDs[16 + lane] = ei[MM + e0 + lane];
  }
  __syncthreads();

  // gather h_final[src], h_final[dst] (float4 loads -> f16 LDS)
  for (int idx = lane; idx < 256; idx += 32) {
    int r = idx >> 4, q = idx & 15;
    v4f a = *(const v4f*)&hfin[(size_t)IDs[r] * 64 + q * 4];
    v4f b = *(const v4f*)&hfin[(size_t)IDs[16 + r] * 64 + q * 4];
    v4h ah, bh;
#pragma unroll
    for (int j = 0; j < 4; ++j) { ah[j] = (half_t)a[j]; bh[j] = (half_t)b[j]; }
    *(v4h*)&HS[r * 64 + q * 4] = ah;
    *(v4h*)&HD[r * 64 + q * 4] = bh;
  }
  for (int idx = lane; idx < 16 * 9; idx += 32) {
    int r = idx / 9, k = idx % 9;
    int e = e0 + r;
    EX[idx] = (k < 5) ? eattr[(size_t)e * 5 + k]
                      : eseq[((size_t)e * KK + (KK - 1)) * 4 + (k - 5)];
  }
  __syncthreads();

  // z1 = relu([h_src|h_dst|extra9] @ W_d1 + b_d1): 4 WMMAs + 9-dim VALU tail
  v16h s0 = a_frag_lds(HS, 64, 0, 0), s1 = a_frag_lds(HS, 64, 0, 32);
  v16h d0 = a_frag_lds(HD, 64, 0, 0), d1 = a_frag_lds(HD, 64, 0, 32);
#pragma unroll
  for (int ct = 0; ct < 4; ++ct) {
    v8f acc = {};
    acc = wmma16(s0, b_frag_t(WL, 128, 0, ct * 16), acc);
    acc = wmma16(s1, b_frag_t(WL, 128, 32, ct * 16), acc);
    acc = wmma16(d0, b_frag_t(WL, 128, 64, ct * 16), acc);
    acc = wmma16(d1, b_frag_t(WL, 128, 96, ct * 16), acc);
    int col = ct * 16 + m;
    float bb = bd1[col];
#pragma unroll
    for (int e = 0; e < 8; ++e) {
      int r = e + hf * 8;
      float s = acc[e] + bb;
#pragma unroll
      for (int k = 0; k < 9; ++k) s += EX[r * 9 + k] * Wd1[(128 + k) * 64 + col];
      Z1[r * 64 + col] = (half_t)(s > 0.f ? s : 0.f);
    }
  }
  __syncthreads();

  // z2 = relu(z1 @ W_d2 + b_d2)   (64 -> 32)
  v16h z0 = a_frag_lds(Z1, 64, 0, 0), z1f = a_frag_lds(Z1, 64, 0, 32);
#pragma unroll
  for (int ct = 0; ct < 2; ++ct) {
    v8f acc = {};
    acc = wmma16(z0, b_frag_t(WL2, 64, 0, ct * 16), acc);
    acc = wmma16(z1f, b_frag_t(WL2, 64, 32, ct * 16), acc);
    int col = ct * 16 + m;
    float bb = bd2[col];
#pragma unroll
    for (int e = 0; e < 8; ++e) {
      float s = acc[e] + bb;
      Z2[(e + hf * 8) * 32 + col] = (half_t)(s > 0.f ? s : 0.f);
    }
  }
  __syncthreads();

  // out = z2 @ W_d3 + b_d3   (32 -> 1)
  if (lane < 16) {
    float s = bd3[0];
#pragma unroll
    for (int k = 0; k < 32; ++k) s += (float)Z2[lane * 32 + k] * Wd3[k];
    out[e0 + lane] = s;
  }
}

// ---------------------------------------------------------------- launch
extern "C" void kernel_launch(void* const* d_in, const int* in_sizes, int n_in,
                              void* d_out, int out_size, void* d_ws, size_t ws_size,
                              hipStream_t stream) {
  const float* x = (const float*)d_in[0];
  const int* ei = (const int*)d_in[1];
  const float* eattr = (const float*)d_in[2];
  const float* u = (const float*)d_in[3];
  const float* eseq = (const float*)d_in[4];
  const float* useq = (const float*)d_in[5];
  const float* W_proj = (const float*)d_in[6];
  const float* b_proj = (const float*)d_in[7];
  const float* W_gcn = (const float*)d_in[8];
  const float* b_gcn = (const float*)d_in[9];
  const float* Wq = (const float*)d_in[10]; const float* bq = (const float*)d_in[11];
  const float* Wk = (const float*)d_in[12]; const float* bk = (const float*)d_in[13];
  const float* Wv = (const float*)d_in[14]; const float* bv = (const float*)d_in[15];
  const float* Wo = (const float*)d_in[16]; const float* bo = (const float*)d_in[17];
  const float* ln1g = (const float*)d_in[18]; const float* ln1b = (const float*)d_in[19];
  const float* Wf1 = (const float*)d_in[20]; const float* bf1 = (const float*)d_in[21];
  const float* Wf2 = (const float*)d_in[22]; const float* bf2 = (const float*)d_in[23];
  const float* ln2g = (const float*)d_in[24]; const float* ln2b = (const float*)d_in[25];
  const float* Wd1 = (const float*)d_in[26]; const float* bd1 = (const float*)d_in[27];
  const float* Wd2 = (const float*)d_in[28]; const float* bd2 = (const float*)d_in[29];
  const float* Wd3 = (const float*)d_in[30]; const float* bd3 = (const float*)d_in[31];

  char* ws = (char*)d_ws;
  size_t off = 0;
  auto alloc = [&](size_t bytes) {
    size_t o = (off + 255) & ~(size_t)255;
    off = o + bytes;
    return o;
  };
  float* dinv = (float*)(ws + alloc(sizeof(float) * NN));
  half_t* xg = (half_t*)(ws + alloc(sizeof(half_t) * (size_t)NN * KK * HH));
  float* agg = (float*)(ws + alloc(sizeof(float) * (size_t)NN * KK * HH));
  float* hfin = (float*)(ws + alloc(sizeof(float) * (size_t)NN * HH));
  half_t* wpT  = (half_t*)(ws + alloc(sizeof(half_t) * 64 * 32));
  half_t* wgT  = (half_t*)(ws + alloc(sizeof(half_t) * 64 * 64));
  half_t* wqT  = (half_t*)(ws + alloc(sizeof(half_t) * 64 * 64));
  half_t* wkT  = (half_t*)(ws + alloc(sizeof(half_t) * 64 * 64));
  half_t* wvT  = (half_t*)(ws + alloc(sizeof(half_t) * 64 * 64));
  half_t* woT  = (half_t*)(ws + alloc(sizeof(half_t) * 64 * 64));
  half_t* wf1T = (half_t*)(ws + alloc(sizeof(half_t) * 128 * 64));
  half_t* wf2T = (half_t*)(ws + alloc(sizeof(half_t) * 64 * 128));
  half_t* wd1T = (half_t*)(ws + alloc(sizeof(half_t) * 64 * 128));
  half_t* wd2T = (half_t*)(ws + alloc(sizeof(half_t) * 32 * 64));

  // fragment-ready transposed f16 weights
  k_transp<<<8, 256, 0, stream>>>(W_proj, wpT, 28, 32, 64);
  k_transp<<<16, 256, 0, stream>>>(W_gcn, wgT, 64, 64, 64);
  k_transp<<<16, 256, 0, stream>>>(Wq, wqT, 64, 64, 64);
  k_transp<<<16, 256, 0, stream>>>(Wk, wkT, 64, 64, 64);
  k_transp<<<16, 256, 0, stream>>>(Wv, wvT, 64, 64, 64);
  k_transp<<<16, 256, 0, stream>>>(Wo, woT, 64, 64, 64);
  k_transp<<<32, 256, 0, stream>>>(Wf1, wf1T, 64, 64, 128);
  k_transp<<<32, 256, 0, stream>>>(Wf2, wf2T, 128, 128, 64);
  k_transp<<<32, 256, 0, stream>>>(Wd1, wd1T, 128, 128, 64);  // rows 0..127
  k_transp<<<8, 256, 0, stream>>>(Wd2, wd2T, 64, 64, 32);

  k_zero_f32<<<(NN + 255) / 256, 256, 0, stream>>>(dinv, NN);
  k_deg<<<MM / 256, 256, 0, stream>>>(ei, dinv);
  k_dinv<<<(NN + 255) / 256, 256, 0, stream>>>(dinv);
  k_proj_gcn<<<(NROWTILES + 7) / 8, 256, 0, stream>>>(x, u, useq, wpT, b_proj, wgT, xg);
  k_init_agg<<<(NN * KK * 64 / 4) / 256, 256, 0, stream>>>(xg, dinv, agg);
  k_scatter<<<MM / 8, 256, 0, stream>>>(ei, dinv, xg, agg);
  k_transformer<<<NN / 16, 64, 0, stream>>>(agg, b_gcn, wqT, bq, wkT, bk, wvT, bv,
                                            woT, bo, ln1g, ln1b, wf1T, bf1, wf2T,
                                            bf2, ln2g, ln2b, hfin);
  k_edge_mlp<<<MM / 64, 128, 0, stream>>>(ei, eattr, eseq, hfin, wd1T, Wd1, bd1,
                                          wd2T, bd2, Wd3, bd3, (float*)d_out);
}
